// LSTMmodel_22119081574889
// MI455X (gfx1250) — matrix operand order repaired and checked by reading, not compiled, not run on hardware
//
#include <hip/hip_runtime.h>
#include <hip/hip_bf16.h>
#include <math.h>

// Problem constants (match reference)
constexpr int BATCH = 1024;
constexpr int TENC  = 384;
constexpr int TDEC  = 128;
constexpr int IDIM  = 128;
constexpr int HDIM  = 512;
constexpr int FOURH = 4 * HDIM;   // 2048

constexpr int ROWLEN = 1032;      // padded LDS row (max KT=1024, +8 keeps 16B chunks aligned, kills bank conflicts)
constexpr int ROWFC  = 520;       // padded LDS row for FC (K=512)

typedef __attribute__((ext_vector_type(16))) __bf16 v16bf;
typedef __attribute__((ext_vector_type(8)))  __bf16 v8bf;
typedef __attribute__((ext_vector_type(8)))  float  v8f;

__device__ __forceinline__ __bf16 f2bf(float f) {
  unsigned u = __builtin_bit_cast(unsigned, f);
  u = (u + 0x7FFFu + ((u >> 16) & 1u)) >> 16;   // round-to-nearest-even
  return __builtin_bit_cast(__bf16, (unsigned short)u);
}

__device__ __forceinline__ float sigmoid_f(float x) {
  return 1.0f / (1.0f + __expf(-x));
}

// Per-lane async global->LDS 16-byte copy (CDNA5, tracked by ASYNCcnt).
// LDS address operand = low 32 bits of the generic pointer (ISA 10.2).
__device__ __forceinline__ void async_copy_b128(const void* gsrc, void* ldst) {
  unsigned lds = (unsigned)(unsigned long long)(uintptr_t)ldst;
  unsigned long long ga = (unsigned long long)(uintptr_t)gsrc;
  asm volatile("global_load_async_to_lds_b128 %0, %1, off" :: "v"(lds), "v"(ga) : "memory");
}
__device__ __forceinline__ void async_wait0() {
  asm volatile("s_wait_asynccnt 0x0" ::: "memory");
}

// Build A fragment in contiguous k order (same order as the B loads):
// lanes<16 -> K {0..15}; lanes>=16 -> K {16..31}.
__device__ __forceinline__ v16bf load_a_frag(const __bf16* p, int halfA) {
  v8bf lo = *(const v8bf*)(p + 2 * halfA);
  v8bf hi = *(const v8bf*)(p + 2 * halfA + 8);
  v16bf a;
#pragma unroll
  for (int i = 0; i < 8; ++i) { a[i] = lo[i]; a[8 + i] = hi[i]; }
  return a;
}

// ---------------------------------------------------------------------------
// Generic LSTM cell step body.  z = [x_t | h] @ [Wih ; Whh]^T + bias; gates.
// Block = 128 threads (4 waves).  Block tile: 32 batch rows x 64 h-columns.
// Each wave: M=32 (2 A-tiles) x 4 gate tiles, B fragment reused across M.
// ---------------------------------------------------------------------------
struct CellArgs {
  const float*  xf;     // fp32 x rows (encoder layer0), or nullptr
  const __bf16* xb;     // bf16 x rows (h0 feed / decoder feedback), or nullptr
  long          xrs;    // x row stride in elements
  int           iin;    // I_IN (128 or 512)
  const __bf16* hbf;    // h state in, BATCH x HDIM bf16
  const __bf16* Wih;    // (4H x I_IN) bf16 row-major
  const __bf16* Whh;    // (4H x HDIM) bf16 row-major
  const float*  bias;   // 4H fp32 (b_ih + b_hh)
  float*        c;      // BATCH x HDIM fp32, in/out
  __bf16*       hout;   // BATCH x HDIM bf16
};

__device__ __forceinline__ void cell_body(const CellArgs& A,
                                          __bf16 (&sA)[32][ROWLEN],
                                          int bt, int cg) {
  const int tid = threadIdx.x;
  const int iin = A.iin;

  // ---- Stage x-part ----
  if (A.xf) {
    // fp32 -> bf16 convert (encoder layer0 only; 32 rows x iin cols, 4 thr/row)
    const int row = tid >> 2, sub = tid & 3;
    const float* xr = A.xf + (size_t)(bt * 32 + row) * A.xrs;
    for (int k = sub * 4; k < iin; k += 16) {
      float4 v = *(const float4*)(xr + k);
      sA[row][k + 0] = f2bf(v.x);
      sA[row][k + 1] = f2bf(v.y);
      sA[row][k + 2] = f2bf(v.z);
      sA[row][k + 3] = f2bf(v.w);
    }
  } else {
    // bf16 direct: async DMA to LDS, 16B chunks
    const int sh  = (iin == 128) ? 4 : 6;     // chunks per row = iin/8
    const int msk = (1 << sh) - 1;
    for (int idx = tid; idx < (32 << sh); idx += 128) {
      const int row = idx >> sh, c8 = idx & msk;
      async_copy_b128(A.xb + (size_t)(bt * 32 + row) * A.xrs + c8 * 8,
                      &sA[row][c8 * 8]);
    }
  }
  // ---- Stage h-part (bf16 async DMA), 32 rows x 64 chunks ----
  for (int idx = tid; idx < 32 * 64; idx += 128) {
    const int row = idx >> 6, c8 = idx & 63;
    async_copy_b128(A.hbf + (size_t)(bt * 32 + row) * HDIM + c8 * 8,
                    &sA[row][iin + c8 * 8]);
  }
  async_wait0();
  __syncthreads();

  // ---- WMMA compute ----
  const int lane  = tid & 31;
  const int wv    = tid >> 5;
  const int n0    = cg * 64 + wv * 16;
  const int n     = lane & 15;
  const int halfA = (lane & 16) ? 8 : 0;
  const int halfB = (lane & 16) ? 16 : 0;

  v8f acc[2][4] = {};
  const __bf16* a0 = &sA[n][0];        // A-tile rows 0..15
  const __bf16* a1 = &sA[16 + n][0];   // A-tile rows 16..31
  const __bf16* wI[4] = { A.Wih + (size_t)(0 * HDIM + n0 + n) * iin,
                          A.Wih + (size_t)(1 * HDIM + n0 + n) * iin,
                          A.Wih + (size_t)(2 * HDIM + n0 + n) * iin,
                          A.Wih + (size_t)(3 * HDIM + n0 + n) * iin };
  const __bf16* wH[4] = { A.Whh + (size_t)(0 * HDIM + n0 + n) * HDIM,
                          A.Whh + (size_t)(1 * HDIM + n0 + n) * HDIM,
                          A.Whh + (size_t)(2 * HDIM + n0 + n) * HDIM,
                          A.Whh + (size_t)(3 * HDIM + n0 + n) * HDIM };

#pragma unroll 2
  for (int k0 = 0; k0 < iin; k0 += 32) {
    v16bf A0 = load_a_frag(a0 + k0, halfA);
    v16bf A1 = load_a_frag(a1 + k0, halfA);
#pragma unroll
    for (int g = 0; g < 4; ++g) {
      v16bf Bv = *(const v16bf*)(wI[g] + k0 + halfB);
      acc[0][g] = __builtin_amdgcn_wmma_f32_16x16x32_bf16(false, A0, false, Bv, (short)0, acc[0][g], false, false);
      acc[1][g] = __builtin_amdgcn_wmma_f32_16x16x32_bf16(false, A1, false, Bv, (short)0, acc[1][g], false, false);
    }
  }
#pragma unroll 2
  for (int k0 = 0; k0 < HDIM; k0 += 32) {
    v16bf A0 = load_a_frag(a0 + iin + k0, halfA);
    v16bf A1 = load_a_frag(a1 + iin + k0, halfA);
#pragma unroll
    for (int g = 0; g < 4; ++g) {
      v16bf Bv = *(const v16bf*)(wH[g] + k0 + halfB);
      acc[0][g] = __builtin_amdgcn_wmma_f32_16x16x32_bf16(false, A0, false, Bv, (short)0, acc[0][g], false, false);
      acc[1][g] = __builtin_amdgcn_wmma_f32_16x16x32_bf16(false, A1, false, Bv, (short)0, acc[1][g], false, false);
    }
  }

  // ---- Elementwise LSTM update (fp32 cell state, bf16 h out) ----
  const int rowoff = (lane & 16) ? 8 : 0;
  const int col    = n0 + n;
  const float bI = A.bias[0 * HDIM + col];
  const float bF = A.bias[1 * HDIM + col];
  const float bG = A.bias[2 * HDIM + col];
  const float bO = A.bias[3 * HDIM + col];
#pragma unroll
  for (int m = 0; m < 2; ++m) {
#pragma unroll
    for (int r = 0; r < 8; ++r) {
      const int brow = bt * 32 + m * 16 + r + rowoff;
      const size_t idx = (size_t)brow * HDIM + col;
      float ig = sigmoid_f(acc[m][0][r] + bI);
      float fg = sigmoid_f(acc[m][1][r] + bF);
      float gg = tanhf(acc[m][2][r] + bG);
      float og = sigmoid_f(acc[m][3][r] + bO);
      float cn = fg * A.c[idx] + ig * gg;
      A.c[idx]    = cn;
      A.hout[idx] = f2bf(og * tanhf(cn));
    }
  }
}

// Encoder step, software-pipelined across layers:
// z=0 -> layer0 step t (if run0), z=1 -> layer1 step t-1 (if run1).
__global__ __launch_bounds__(128)
void enc_step(CellArgs L0, CellArgs L1, int run0, int run1) {
  __shared__ __attribute__((aligned(32))) __bf16 sA[32][ROWLEN];
  if (blockIdx.z == 0) {
    if (!run0) return;
    cell_body(L0, sA, blockIdx.x, blockIdx.y);
  } else {
    if (!run1) return;
    cell_body(L1, sA, blockIdx.x, blockIdx.y);
  }
}

// Standalone cell (decoder stages).
__global__ __launch_bounds__(128)
void cell_kernel(CellArgs A) {
  __shared__ __attribute__((aligned(32))) __bf16 sA[32][ROWLEN];
  cell_body(A, sA, blockIdx.x, blockIdx.y);
}

// ---------------------------------------------------------------------------
// FC readout: out = h1 @ fc_w^T + fc_b (M=1024, N=128, K=512) via WMMA.
// Writes fp32 prediction to d_out[:, t, :] and bf16 feedback for the decoder.
// Grid: (BATCH/32, IDIM/64), block 128.
// ---------------------------------------------------------------------------
__global__ __launch_bounds__(128)
void fc_wmma(const __bf16* __restrict__ h_in,   // BATCH x HDIM bf16
             const __bf16* __restrict__ Wfc,    // (IDIM x HDIM) bf16 row-major
             const float* __restrict__ fcb,     // IDIM fp32
             float* __restrict__ out_base,      // d_out + t*IDIM
             long out_rs,                       // TDEC*IDIM
             __bf16* __restrict__ dec_bf)       // BATCH x IDIM bf16 feedback
{
  __shared__ __attribute__((aligned(32))) __bf16 sA[32][ROWFC];
  const int tid = threadIdx.x;
  const int bt  = blockIdx.x;
  const int cg  = blockIdx.y;

  for (int idx = tid; idx < 32 * 64; idx += 128) {
    const int row = idx >> 6, c8 = idx & 63;
    async_copy_b128(h_in + (size_t)(bt * 32 + row) * HDIM + c8 * 8,
                    &sA[row][c8 * 8]);
  }
  async_wait0();
  __syncthreads();

  const int lane  = tid & 31;
  const int wv    = tid >> 5;
  const int n0    = cg * 64 + wv * 16;
  const int n     = lane & 15;
  const int halfA = (lane & 16) ? 8 : 0;
  const int halfB = (lane & 16) ? 16 : 0;

  v8f acc[2] = {};
  const __bf16* a0 = &sA[n][0];
  const __bf16* a1 = &sA[16 + n][0];
  const __bf16* wr = Wfc + (size_t)(n0 + n) * HDIM;

#pragma unroll 2
  for (int k0 = 0; k0 < HDIM; k0 += 32) {
    v16bf A0 = load_a_frag(a0 + k0, halfA);
    v16bf A1 = load_a_frag(a1 + k0, halfA);
    v16bf Bv = *(const v16bf*)(wr + k0 + halfB);
    acc[0] = __builtin_amdgcn_wmma_f32_16x16x32_bf16(false, A0, false, Bv, (short)0, acc[0], false, false);
    acc[1] = __builtin_amdgcn_wmma_f32_16x16x32_bf16(false, A1, false, Bv, (short)0, acc[1], false, false);
  }

  const int rowoff = (lane & 16) ? 8 : 0;
  const int col    = n0 + n;
  const float bb   = fcb[col];
#pragma unroll
  for (int m = 0; m < 2; ++m) {
#pragma unroll
    for (int r = 0; r < 8; ++r) {
      const int brow = bt * 32 + m * 16 + r + rowoff;
      float v = acc[m][r] + bb;
      out_base[(size_t)brow * out_rs + col] = v;
      dec_bf[(size_t)brow * IDIM + col]     = f2bf(v);
    }
  }
}

// ---------------------------------------------------------------------------
// Small utility kernels
// ---------------------------------------------------------------------------
__global__ void k_f32_to_bf16(const float* __restrict__ src, __bf16* __restrict__ dst, int n) {
  int i = blockIdx.x * blockDim.x + threadIdx.x;
  if (i < n) dst[i] = f2bf(src[i]);
}
__global__ void k_bias_sum(const float* __restrict__ a, const float* __restrict__ b,
                           float* __restrict__ dst, int n) {
  int i = blockIdx.x * blockDim.x + threadIdx.x;
  if (i < n) dst[i] = a[i] + b[i];
}
__global__ void k_zero_f32(float* __restrict__ p, int n) {
  int i = blockIdx.x * blockDim.x + threadIdx.x;
  if (i < n) p[i] = 0.0f;
}
__global__ void k_zero_bf16(__bf16* __restrict__ p, int n) {
  int i = blockIdx.x * blockDim.x + threadIdx.x;
  if (i < n) p[i] = (__bf16)0.0f;
}
// Extract x[:, TENC-1, :] as bf16 (decoder step-0 input).
__global__ void k_xlast(const float* __restrict__ x, __bf16* __restrict__ dst) {
  int i = blockIdx.x * blockDim.x + threadIdx.x;
  if (i < BATCH * IDIM) {
    int b = i >> 7, k = i & 127;
    dst[i] = f2bf(x[(size_t)b * TENC * IDIM + (size_t)(TENC - 1) * IDIM + k]);
  }
}

// ---------------------------------------------------------------------------
extern "C" void kernel_launch(void* const* d_in, const int* in_sizes, int n_in,
                              void* d_out, int out_size, void* d_ws, size_t ws_size,
                              hipStream_t stream) {
  (void)in_sizes; (void)n_in; (void)out_size; (void)ws_size;

  const float* x     = (const float*)d_in[0];   // (B, TENC, I)
  const float* W_ih0 = (const float*)d_in[1];   // (4H, I)
  const float* W_hh0 = (const float*)d_in[2];   // (4H, H)
  const float* b_ih0 = (const float*)d_in[3];
  const float* b_hh0 = (const float*)d_in[4];
  const float* W_ih1 = (const float*)d_in[5];   // (4H, H)
  const float* W_hh1 = (const float*)d_in[6];   // (4H, H)
  const float* b_ih1 = (const float*)d_in[7];
  const float* b_hh1 = (const float*)d_in[8];
  const float* fc_w  = (const float*)d_in[9];   // (I, H)
  const float* fc_b  = (const float*)d_in[10];
  float* out = (float*)d_out;                   // (B, TDEC, I)

  // Workspace carve-up (256B aligned regions)
  char* p = (char*)d_ws;
  auto carve = [&](size_t bytes) -> void* {
    void* r = (void*)p;
    p += (bytes + 255) & ~(size_t)255;
    return r;
  };
  __bf16* Wih0b = (__bf16*)carve((size_t)FOURH * IDIM * 2);
  __bf16* Whh0b = (__bf16*)carve((size_t)FOURH * HDIM * 2);
  __bf16* Wih1b = (__bf16*)carve((size_t)FOURH * HDIM * 2);
  __bf16* Whh1b = (__bf16*)carve((size_t)FOURH * HDIM * 2);
  __bf16* Wfcb  = (__bf16*)carve((size_t)IDIM * HDIM * 2);
  float*  b0    = (float*)carve((size_t)FOURH * 4);
  float*  b1    = (float*)carve((size_t)FOURH * 4);
  __bf16* h0A   = (__bf16*)carve((size_t)BATCH * HDIM * 2);
  __bf16* h0B   = (__bf16*)carve((size_t)BATCH * HDIM * 2);
  __bf16* h1A   = (__bf16*)carve((size_t)BATCH * HDIM * 2);
  __bf16* h1B   = (__bf16*)carve((size_t)BATCH * HDIM * 2);
  float*  c0    = (float*)carve((size_t)BATCH * HDIM * 4);
  float*  c1    = (float*)carve((size_t)BATCH * HDIM * 4);
  __bf16* decbf = (__bf16*)carve((size_t)BATCH * IDIM * 2);
  __bf16* xlast = (__bf16*)carve((size_t)BATCH * IDIM * 2);

  auto g1 = [](int n) { return dim3((n + 255) / 256); };
  k_f32_to_bf16<<<g1(FOURH * IDIM), 256, 0, stream>>>(W_ih0, Wih0b, FOURH * IDIM);
  k_f32_to_bf16<<<g1(FOURH * HDIM), 256, 0, stream>>>(W_hh0, Whh0b, FOURH * HDIM);
  k_f32_to_bf16<<<g1(FOURH * HDIM), 256, 0, stream>>>(W_ih1, Wih1b, FOURH * HDIM);
  k_f32_to_bf16<<<g1(FOURH * HDIM), 256, 0, stream>>>(W_hh1, Whh1b, FOURH * HDIM);
  k_f32_to_bf16<<<g1(IDIM * HDIM),  256, 0, stream>>>(fc_w,  Wfcb,  IDIM * HDIM);
  k_bias_sum<<<g1(FOURH), 256, 0, stream>>>(b_ih0, b_hh0, b0, FOURH);
  k_bias_sum<<<g1(FOURH), 256, 0, stream>>>(b_ih1, b_hh1, b1, FOURH);
  k_zero_bf16<<<g1(BATCH * HDIM), 256, 0, stream>>>(h0A, BATCH * HDIM);
  k_zero_bf16<<<g1(BATCH * HDIM), 256, 0, stream>>>(h1A, BATCH * HDIM);
  k_zero_f32<<<g1(BATCH * HDIM), 256, 0, stream>>>(c0, BATCH * HDIM);
  k_zero_f32<<<g1(BATCH * HDIM), 256, 0, stream>>>(c1, BATCH * HDIM);
  k_xlast<<<g1(BATCH * IDIM), 256, 0, stream>>>(x, xlast);

  dim3 blk(128);
  dim3 genc(BATCH / 32, HDIM / 64, 2);
  dim3 gcell(BATCH / 32, HDIM / 64);
  dim3 gfc(BATCH / 32, IDIM / 64);

  __bf16* h0c = h0A; __bf16* h0n = h0B;
  __bf16* h1c = h1A; __bf16* h1n = h1B;

  // ---- Encoder: one launch per step; z=0 layer0(t), z=1 layer1(t-1) ----
  for (int t = 0; t <= TENC; ++t) {
    const int run0 = (t < TENC), run1 = (t >= 1);
    const int tx = run0 ? t : (TENC - 1);
    CellArgs L0 = { x + (size_t)tx * IDIM, nullptr, (long)TENC * IDIM, IDIM,
                    h0c, Wih0b, Whh0b, b0, c0, h0n };
    CellArgs L1 = { nullptr, h0c, (long)HDIM, HDIM,
                    h1c, Wih1b, Whh1b, b1, c1, h1n };
    enc_step<<<genc, blk, 0, stream>>>(L0, L1, run0, run1);
    if (run0) { __bf16* t0 = h0c; h0c = h0n; h0n = t0; }
    if (run1) { __bf16* t1 = h1c; h1c = h1n; h1n = t1; }
  }

  // ---- Decoder: cell0 -> cell1 -> FC (serial through the feedback) ----
  for (int t = 0; t < TDEC; ++t) {
    CellArgs D0 = { nullptr, (t == 0) ? xlast : decbf, (long)IDIM, IDIM,
                    h0c, Wih0b, Whh0b, b0, c0, h0n };
    cell_kernel<<<gcell, blk, 0, stream>>>(D0);
    CellArgs D1 = { nullptr, h0n, (long)HDIM, HDIM,
                    h1c, Wih1b, Whh1b, b1, c1, h1n };
    cell_kernel<<<gcell, blk, 0, stream>>>(D1);
    fc_wmma<<<gfc, blk, 0, stream>>>(h1n, Wfcb, fc_b,
                                     out + (size_t)t * IDIM, (long)TDEC * IDIM, decbf);
    __bf16* t0 = h0c; h0c = h0n; h0n = t0;
    __bf16* t1 = h1c; h1c = h1n; h1n = t1;
  }
}
